// QuantumGraphConvNet_53807350284467
// MI455X (gfx1250) — compile-verified, matrix operand order checked by reading, not run on hardware
//
#include <hip/hip_runtime.h>
#include <math.h>
#include <stdint.h>

// ============================================================================
// QuantumGraphConvNet forward for MI455X (gfx1250), wave32 + WMMA + TDM.
//
// Key transformations vs the reference:
//  * herm_sqrt/eigh are dead code (U_dil[:K2,:K2] == U)  -> no eigensolver.
//  * per-edge complex matmul reassociated: scatter first, then one GEMM.
//  * expm(A) = scaling-and-squaring (s=6) + order-12 Taylor (Horner):
//      17 complex 656^3 GEMMs, each = 4 real GEMMs on v_wmma_f32_16x16x32_f16
//      with an f32 = f16_hi + f16_lo operand split (3 WMMAs / k-tile) for
//      ~2^-22 product accuracy.
//  * Operands are PRE-PACKED into the exact WMMA wave32 fragment layout
//    (one v16h per lane per tile): the f32->f16 split happens once per GEMM;
//    the GEMM inner loop is pure loads + v_wmma + global_prefetch_b8.
//    X (Horner A-operand) is packed once for all 11 iterations.
//  * Complex GEMMs run as 8-wave workgroups sharing the A fragment panel in
//    LDS (42KB of 320KB/WGP): the panel is DMA'd by the Tensor Data Mover
//    (tensor_load_to_lds + s_wait_tensorcnt), cutting A global traffic 8x;
//    A-fragment reads become ds_load_b128.
//  * TopK pooling: exact JAX top_k ordering via in-LDS bitonic sort.
// ============================================================================

#define N0_    4096
#define EDG    65536
#define DIN_   128
#define H_     256
#define NC_    10
#define UH_    64
#define UH2_   32
#define K1_    3277
#define K2_    656
#define K2SQ_  (K2_ * K2_)      // 430336
#define SORTN  4096
#define EXPM_M 12
#define EXPM_S 6
#define KT_MAX 21               // max K/32 tiles in the LDS-staged GEMM (656/32)

// pack buffer capacities (in v16h fragments); v16h = 32 bytes = 8 f32 slots
#define APACK_V 57344           // >= max MT*KT*32 (gcn2: 205*8*32 = 52480)
#define BPACK_V 28672           // >= max NT*KT*32 (expm: 41*21*32 = 27552)

typedef __attribute__((ext_vector_type(16))) _Float16     v16h;
typedef __attribute__((ext_vector_type(8)))  float        v8f;
typedef __attribute__((ext_vector_type(4)))  unsigned int u32x4;
typedef __attribute__((ext_vector_type(8)))  int          i32x8;
typedef __attribute__((ext_vector_type(4)))  int          i32x4;

struct HiLo { _Float16 hi, lo; };
__device__ __forceinline__ HiLo splitf(float x) {
  _Float16 h = (_Float16)x;
  HiLo r; r.hi = h; r.lo = (_Float16)(x - (float)h);
  return r;
}

// ---------------------------------------------------------------------------
// Tensor Data Mover: 1-D contiguous copy global -> LDS (ISA ch.8 D# layout).
// group0: [1:0]=count=1 | lds_addr | global_addr[56:0] | type=2 ("image")
// group1: data_size=4B; tensor_dim0 = tile_dim0 = ndwords; dims 1..4 trivial.
// Issued once per wave (TDM ignores EXEC); completion via TENSORcnt.
// ---------------------------------------------------------------------------
#if __has_builtin(__builtin_amdgcn_tensor_load_to_lds)
#define HAVE_TDM 1
__device__ __forceinline__ void tdm_load_1d(unsigned lds_off, const void* gsrc,
                                            unsigned ndwords)
{
  unsigned long long ga = (unsigned long long)(uintptr_t)gsrc;
  u32x4 g0 = { 1u,                                   // count=1, user mode
               lds_off,                               // lds_addr (bytes)
               (unsigned)ga,                          // global_addr[31:0]
               ((unsigned)(ga >> 32) & 0x01FFFFFFu) | (2u << 30) }; // [56:32]|type=2
  i32x8 g1 = { (int)(2u << 16),                      // data_size = 4B
               (int)((ndwords & 0xFFFFu) << 16),     // tensor_dim0[15:0] @w1[31:16]
               (int)(((ndwords >> 16) & 0xFFFFu) | (1u << 16)), // dim0 hi | tensor_dim1=1
               (int)((ndwords & 0xFFFFu) << 16),     // tile_dim0 @w3[31:16]
               1,                                    // tile_dim1=1, tile_dim2=0
               (int)ndwords,                         // tensor_dim0_stride[31:0]
               0, 0 };
  i32x4 g2 = { 1, 0, 0, 0 };                         // tensor_dim2=1
  i32x4 g3 = { 0, 0, 0, 0 };
#if __clang_major__ >= 23
  i32x8 gpad = {};
  __builtin_amdgcn_tensor_load_to_lds(g0, g1, g2, g3, gpad, 0);
#else
  __builtin_amdgcn_tensor_load_to_lds(g0, g1, g2, g3, 0);
#endif
}
#else
#define HAVE_TDM 0
#endif

// ---------------------------------------------------------------------------
// Fragment packing: f32 matrix -> per-lane v16h fragments (hi/lo split) in
// the ISA wave32 WMMA operand layout. Zero-fills M/K edges.
//   A (MxK, row-major):  tile (tm,tk) 16x32; lanes 0-15 rows, kgrp = lane>>4.
//   B (KxN, row-major):  tile (tn,tk) 32x16; lane col, same K pattern.
// Fragment index = (tileMajor*KT + tk)*32 + lane  (contiguous in tk).
// ---------------------------------------------------------------------------
__global__ void k_pack_a(const float* __restrict__ A, int lda, int M, int K, int KT,
                         v16h* __restrict__ Phi, v16h* __restrict__ Plo, int ntot)
{
  int gid = blockIdx.x * blockDim.x + threadIdx.x;
  if (gid >= ntot) return;
  int lane = gid & 31;
  int tile = gid >> 5;
  int tk = tile % KT, tm = tile / KT;
  int mrow = tm * 16 + (lane & 15);
  int kgrp = (lane >> 4) & 1;
  v16h hi = {}, lo = {};
  if (mrow < M) {
#pragma unroll
    for (int i = 0; i < 8; ++i) {
      int kk = tk * 32 + ((i & 4) << 2) + kgrp * 8 + 2 * (i & 3);
      if (kk < K) {                         // K even -> kk, kk+1 both valid
        const float* ap = A + (size_t)mrow * lda + kk;
        HiLo s0 = splitf(ap[0]), s1 = splitf(ap[1]);
        hi[2 * i] = s0.hi;     lo[2 * i] = s0.lo;
        hi[2 * i + 1] = s1.hi; lo[2 * i + 1] = s1.lo;
      }
    }
  }
  Phi[gid] = hi;
  Plo[gid] = lo;
}

__global__ void k_pack_b(const float* __restrict__ B, int ldb, int K, int N, int KT,
                         v16h* __restrict__ Phi, v16h* __restrict__ Plo, int ntot)
{
  int gid = blockIdx.x * blockDim.x + threadIdx.x;
  if (gid >= ntot) return;
  int lane = gid & 31;
  int tile = gid >> 5;
  int tk = tile % KT, tn = tile / KT;
  int ncol = tn * 16 + (lane & 15);
  int kgrp = (lane >> 4) & 1;
  v16h hi = {}, lo = {};
  if (ncol < N) {
#pragma unroll
    for (int i = 0; i < 8; ++i) {
      int kk = tk * 32 + ((i & 4) << 2) + kgrp * 8 + 2 * (i & 3);
      if (kk < K) {
        const float* bp = B + (size_t)kk * ldb + ncol;
        HiLo s0 = splitf(bp[0]), s1 = splitf(bp[ldb]);
        hi[2 * i] = s0.hi;     lo[2 * i] = s0.lo;
        hi[2 * i + 1] = s1.hi; lo[2 * i + 1] = s1.lo;
      }
    }
  }
  Phi[gid] = hi;
  Plo[gid] = lo;
}

// ---------------------------------------------------------------------------
// WMMA GEMM on pre-packed fragments (per-wave): C = alpha*A*B+beta*C(+bias)(act)
// One wave per 16x16 C tile. Inner loop: 4x 32B fragment loads + 3 WMMAs.
// ---------------------------------------------------------------------------
__global__ __launch_bounds__(32)
void k_gemm_pk(const v16h* __restrict__ Ahi, const v16h* __restrict__ Alo,
               const v16h* __restrict__ Bhi, const v16h* __restrict__ Blo,
               float* __restrict__ C, int ldc, int M, int N, int KT,
               float alpha, float beta, const float* __restrict__ bias, int act)
{
  const int lane = threadIdx.x;            // full wave, EXEC all ones
  const int tm = blockIdx.y, tn = blockIdx.x;
  const size_t ab = ((size_t)tm * KT) * 32 + lane;
  const size_t bb = ((size_t)tn * KT) * 32 + lane;
  v8f acc0 = {}, acc1 = {};
#pragma unroll 2
  for (int tk = 0; tk < KT; ++tk) {
    if (tk + 1 < KT) {                     // gfx1250 global_prefetch_b8
      __builtin_prefetch(&Ahi[ab + (size_t)(tk + 1) * 32]);
      __builtin_prefetch(&Bhi[bb + (size_t)(tk + 1) * 32]);
    }
    v16h a_hi = Ahi[ab + (size_t)tk * 32];
    v16h a_lo = Alo[ab + (size_t)tk * 32];
    v16h b_hi = Bhi[bb + (size_t)tk * 32];
    v16h b_lo = Blo[bb + (size_t)tk * 32];
    // (Ah+Al)(Bh+Bl) ~= AhBh + AhBl + AlBh   (AlBl ~ 2^-22, dropped)
    acc0 = __builtin_amdgcn_wmma_f32_16x16x32_f16(false, a_hi, false, b_hi, (short)0, acc0, false, false);
    acc1 = __builtin_amdgcn_wmma_f32_16x16x32_f16(false, a_hi, false, b_lo, (short)0, acc1, false, false);
    acc1 = __builtin_amdgcn_wmma_f32_16x16x32_f16(false, a_lo, false, b_hi, (short)0, acc1, false, false);
  }
  const int ncol = tn * 16 + (lane & 15);
  const int kgrp = (lane >> 4) & 1;
  if (ncol < N) {
    const float bv = bias ? bias[ncol] : 0.f;
#pragma unroll
    for (int r = 0; r < 8; ++r) {
      const int row = tm * 16 + kgrp * 8 + r;  // C layout: VGPR r -> M=8*kgrp+r
      if (row < M) {
        float* cp = C + (size_t)row * ldc + ncol;
        float v = alpha * (acc0[r] + acc1[r]) + bv;
        if (beta != 0.f) v += beta * cp[0];
        if (act == 1) v = fmaxf(v, 0.f);
        cp[0] = v;
      }
    }
  }
}

// ---------------------------------------------------------------------------
// LDS-staged WMMA GEMM (complex-GEMM workhorse): 8 waves/block share one A
// fragment panel (16 x K strip, hi+lo <= 42KB) staged into LDS by the TDM;
// each wave handles one tn. Requires KT <= KT_MAX.
// ---------------------------------------------------------------------------
__global__ __launch_bounds__(256)
void k_gemm_pk_lds(const v16h* __restrict__ Ahi, const v16h* __restrict__ Alo,
                   const v16h* __restrict__ Bhi, const v16h* __restrict__ Blo,
                   float* __restrict__ C, int ldc, int M, int N, int KT,
                   float alpha, float beta)
{
  __shared__ v16h sAhi[KT_MAX * 32];
  __shared__ v16h sAlo[KT_MAX * 32];
  const int lane = threadIdx.x & 31;
  const int wv   = threadIdx.x >> 5;       // 0..7
  const int tm   = blockIdx.y;
  const int NT   = (N + 15) / 16;
  const int tnq  = blockIdx.x * 8 + wv;
  const int tn   = tnq < NT ? tnq : NT - 1; // clamp; stores guarded below

#if HAVE_TDM
  if (threadIdx.x < 32) {                  // wave 0 issues two TDM DMAs
    tdm_load_1d((unsigned)(uintptr_t)&sAhi[0],
                Ahi + (size_t)tm * KT * 32, (unsigned)KT * 256u);
    tdm_load_1d((unsigned)(uintptr_t)&sAlo[0],
                Alo + (size_t)tm * KT * 32, (unsigned)KT * 256u);
    __builtin_amdgcn_s_wait_tensorcnt(0);
  }
#else
  for (int i = threadIdx.x; i < KT * 32; i += 256) {
    sAhi[i] = Ahi[(size_t)tm * KT * 32 + i];
    sAlo[i] = Alo[(size_t)tm * KT * 32 + i];
  }
#endif
  __syncthreads();

  const size_t bb = ((size_t)tn * KT) * 32 + lane;
  v8f acc0 = {}, acc1 = {};
#pragma unroll 2
  for (int tk = 0; tk < KT; ++tk) {
    if (tk + 1 < KT) __builtin_prefetch(&Bhi[bb + (size_t)(tk + 1) * 32]);
    v16h a_hi = sAhi[tk * 32 + lane];      // ds_load_b128 x2
    v16h a_lo = sAlo[tk * 32 + lane];
    v16h b_hi = Bhi[bb + (size_t)tk * 32];
    v16h b_lo = Blo[bb + (size_t)tk * 32];
    acc0 = __builtin_amdgcn_wmma_f32_16x16x32_f16(false, a_hi, false, b_hi, (short)0, acc0, false, false);
    acc1 = __builtin_amdgcn_wmma_f32_16x16x32_f16(false, a_hi, false, b_lo, (short)0, acc1, false, false);
    acc1 = __builtin_amdgcn_wmma_f32_16x16x32_f16(false, a_lo, false, b_hi, (short)0, acc1, false, false);
  }
  const int ncol = tn * 16 + (lane & 15);
  const int kgrp = (lane >> 4) & 1;
  if (tnq < NT && ncol < N) {
#pragma unroll
    for (int r = 0; r < 8; ++r) {
      const int row = tm * 16 + kgrp * 8 + r;
      if (row < M) {
        float* cp = C + (size_t)row * ldc + ncol;
        float v = alpha * (acc0[r] + acc1[r]);
        if (beta != 0.f) v += beta * cp[0];
        cp[0] = v;
      }
    }
  }
}

// ---------------------------------------------------------------------------
// Small utility kernels
// ---------------------------------------------------------------------------
__global__ void k_fill(float* __restrict__ p, float v, int n) {
  int i = blockIdx.x * blockDim.x + threadIdx.x;
  if (i < n) p[i] = v;
}
__global__ void k_relu(float* __restrict__ p, int n) {
  int i = blockIdx.x * blockDim.x + threadIdx.x;
  if (i < n) p[i] = fmaxf(p[i], 0.f);
}
__global__ void k_scale(const float* __restrict__ s, float* __restrict__ d, float a, int n) {
  int i = blockIdx.x * blockDim.x + threadIdx.x;
  if (i < n) d[i] = a * s[i];
}
__global__ void k_addvec(const float* __restrict__ a, const float* __restrict__ b,
                         float* __restrict__ o, int n) {
  int i = blockIdx.x * blockDim.x + threadIdx.x;
  if (i < n) o[i] = a[i] + b[i];
}
__global__ void k_add_diag(float* __restrict__ M, int n, float v) {
  int i = blockIdx.x * blockDim.x + threadIdx.x;
  if (i < n) M[(size_t)i * n + i] += v;
}
__global__ void k_broadcast_row(const float* __restrict__ w, float* __restrict__ o, int n, int F) {
  int i = blockIdx.x * blockDim.x + threadIdx.x;
  if (i < n * F) o[i] = w[i % F];
}

// deg[dst] += em  (deg prefilled with 1.0 for the self-loop)
__global__ void k_deg(const int* __restrict__ dst, const float* __restrict__ em,
                      float* __restrict__ deg, int E) {
  int e = blockIdx.x * blockDim.x + threadIdx.x;
  if (e >= E) return;
  float m = em ? em[e] : 1.f;
  if (m != 0.f) atomicAdd(&deg[dst[e]], m);
}

// out = h/deg (self-loop: dis^2 * h) + bias
__global__ void k_self_init(const float* __restrict__ h, const float* __restrict__ deg,
                            const float* __restrict__ bias, float* __restrict__ out,
                            int n, int F) {
  int i = blockIdx.x * blockDim.x + threadIdx.x;
  if (i >= n * F) return;
  out[i] = h[i] / deg[i / F] + bias[i % F];
}

// out[dst] += em * rsqrt(deg[src]) * rsqrt(deg[dst]) * h[src]
__global__ void k_gcn_scatter(const float* __restrict__ h, const int* __restrict__ src,
                              const int* __restrict__ dst, const float* __restrict__ em,
                              const float* __restrict__ deg, float* __restrict__ out,
                              int E, int F) {
  int i = blockIdx.x * blockDim.x + threadIdx.x;
  if (i >= E * F) return;
  int e = i / F, f = i % F;
  float m = em ? em[e] : 1.f;
  if (m == 0.f) return;
  int s = src[e], d = dst[e];
  float coef = m * rsqrtf(deg[s]) * rsqrtf(deg[d]);
  atomicAdd(&out[(size_t)d * F + f], coef * h[(size_t)s * F + f]);
}

// out[dst] += em * h[src]   (reassociated message aggregation)
__global__ void k_scatter_plain(const float* __restrict__ h, const int* __restrict__ src,
                                const int* __restrict__ dst, const float* __restrict__ em,
                                float* __restrict__ out, int E, int F) {
  int i = blockIdx.x * blockDim.x + threadIdx.x;
  if (i >= E * F) return;
  int e = i / F, f = i % F;
  float m = em[e];
  if (m == 0.f) return;
  atomicAdd(&out[(size_t)dst[e] * F + f], m * h[(size_t)src[e] * F + f]);
}

// score[i] = dot(h[i], w) / ||w||
__global__ void k_score(const float* __restrict__ h, const float* __restrict__ w,
                        float* __restrict__ sc, int n, int F) {
  int i = blockIdx.x * blockDim.x + threadIdx.x;
  if (i >= n) return;
  float nrm = 0.f, d = 0.f;
  for (int f = 0; f < F; ++f) nrm += w[f] * w[f];
  for (int f = 0; f < F; ++f) d += h[(size_t)i * F + f] * w[f];
  sc[i] = d / sqrtf(nrm);
}

// Full bitonic sort of SORTN (key desc, idx asc on ties == jax.lax.top_k order)
// in 32KB LDS; writes perm[0..k) and mapping[0..n_valid).
__global__ void k_topk_sort(const float* __restrict__ score, int n_valid,
                            int* __restrict__ perm, int* __restrict__ mapping, int k) {
  __shared__ float sk[SORTN];
  __shared__ int   sv[SORTN];
  const int tid = threadIdx.x, nt = blockDim.x;
  for (int i = tid; i < SORTN; i += nt) {
    if (i < n_valid) { sk[i] = score[i]; sv[i] = i; }
    else             { sk[i] = -3.402823466e+38f; sv[i] = 0x7fffffff; }
  }
  __syncthreads();
  for (int kk = 2; kk <= SORTN; kk <<= 1) {
    for (int j = kk >> 1; j > 0; j >>= 1) {
      for (int i = tid; i < SORTN; i += nt) {
        int ixj = i ^ j;
        if (ixj > i) {
          float ka = sk[i], kb = sk[ixj];
          int   va = sv[i], vb = sv[ixj];
          bool b_first = (kb > ka) || (kb == ka && vb < va);
          bool a_first = (ka > kb) || (ka == kb && va < vb);
          bool dir = ((i & kk) == 0);
          if (dir ? b_first : a_first) {
            sk[i] = kb; sk[ixj] = ka; sv[i] = vb; sv[ixj] = va;
          }
        }
      }
      __syncthreads();
    }
  }
  for (int i = tid; i < n_valid; i += nt) mapping[i] = -1;
  for (int i = tid; i < k; i += nt) perm[i] = sv[i];
  __syncthreads();
  for (int i = tid; i < k; i += nt) mapping[sv[i]] = i;
}

// x_new[i] = h[perm[i]] * tanh(score[perm[i]])
__global__ void k_gather_scale(const float* __restrict__ h, const int* __restrict__ perm,
                               const float* __restrict__ sc, float* __restrict__ out,
                               int k, int F) {
  int i = blockIdx.x * blockDim.x + threadIdx.x;
  if (i >= k * F) return;
  int p = perm[i / F];
  out[i] = h[(size_t)p * F + (i % F)] * tanhf(sc[p]);
}

// edge remap after pooling (safe in-place: one thread per edge)
__global__ void k_remap(const int* __restrict__ si, const int* __restrict__ di,
                        const float* __restrict__ emi, const int* __restrict__ mapping,
                        int* __restrict__ so, int* __restrict__ dso,
                        float* __restrict__ emo, int E) {
  int e = blockIdx.x * blockDim.x + threadIdx.x;
  if (e >= E) return;
  int ns = mapping[si[e]], nd = mapping[di[e]];
  float m = emi ? emi[e] : 1.f;
  emo[e] = (ns >= 0 && nd >= 0) ? m : 0.f;
  so[e]  = ns > 0 ? ns : 0;
  dso[e] = nd > 0 ? nd : 0;
}

// y[j] = act(sum_k x[k]*W[k*N+j] + b[j]); act: 0 none, 1 relu, 2 tanh
__global__ void k_vecmat(const float* __restrict__ x, const float* __restrict__ W,
                         const float* __restrict__ b, float* __restrict__ y,
                         int K, int N, int act) {
  int j = blockIdx.x * blockDim.x + threadIdx.x;
  if (j >= N) return;
  float s = b ? b[j] : 0.f;
  for (int k = 0; k < K; ++k) s += x[k] * W[(size_t)k * N + j];
  if (act == 1) s = fmaxf(s, 0.f);
  else if (act == 2) s = tanhf(s);
  y[j] = s;
}

// X = (A/2^s): Xr = sc*(pm - pm^T), Xi = sc*(pm + pm^T), sc = 0.05*0.5/2^s
__global__ void k_buildX(const float* __restrict__ pm, float* __restrict__ Xr,
                         float* __restrict__ Xi, int n, float sc) {
  int i = blockIdx.x * blockDim.x + threadIdx.x;
  if (i >= n * n) return;
  int r = i / n, c = i % n;
  float d = pm[i], dT = pm[(size_t)c * n + r];
  Xr[i] = sc * (d - dT);
  Xi[i] = sc * (d + dT);
}

// outc = agg + (xr + i*xi); mag = |outc|; ph = atan2(im, re)
__global__ void k_polar(const float* __restrict__ ar, const float* __restrict__ ai,
                        const float* __restrict__ xr, const float* __restrict__ xi,
                        float* __restrict__ mag, float* __restrict__ ph, int n) {
  int i = blockIdx.x * blockDim.x + threadIdx.x;
  if (i >= n) return;
  float re = ar[i] + xr[i], im = ai[i] + xi[i];
  mag[i] = sqrtf(re * re + im * im);
  ph[i]  = atan2f(im, re);
}

__global__ void k_colmean(const float* __restrict__ x, float* __restrict__ y, int n, int F) {
  int j = blockIdx.x * blockDim.x + threadIdx.x;
  if (j >= F) return;
  float s = 0.f;
  for (int i = 0; i < n; ++i) s += x[(size_t)i * F + j];
  y[j] = s / (float)n;
}

// ---------------------------------------------------------------------------
extern "C" void kernel_launch(void* const* d_in, const int* in_sizes, int n_in,
                              void* d_out, int out_size, void* d_ws, size_t ws_size,
                              hipStream_t stream)
{
  (void)in_sizes; (void)n_in; (void)out_size; (void)ws_size;
  const float* x       = (const float*)d_in[0];
  const int*   e_src   = (const int*)d_in[1];
  const int*   e_dst   = ((const int*)d_in[1]) + EDG;
  const float* gcn1_w  = (const float*)d_in[2];
  const float* gcn1_b  = (const float*)d_in[3];
  const float* pool1_w = (const float*)d_in[4];
  const float* gcn2_w  = (const float*)d_in[5];
  const float* gcn2_b  = (const float*)d_in[6];
  const float* pool2_w = (const float*)d_in[7];
  const float* inr_w   = (const float*)d_in[8];
  const float* inr_b   = (const float*)d_in[9];
  const float* ini_w   = (const float*)d_in[10];
  const float* ini_b   = (const float*)d_in[11];
  const float* enc1_w  = (const float*)d_in[12];
  const float* enc1_b  = (const float*)d_in[13];
  const float* enc2_w  = (const float*)d_in[14];
  const float* enc2_b  = (const float*)d_in[15];
  const float* p1_w    = (const float*)d_in[16];
  const float* p1_b    = (const float*)d_in[17];
  const float* p2_w    = (const float*)d_in[18];
  const float* p2_b    = (const float*)d_in[19];
  const float* p3_w    = (const float*)d_in[20];
  const float* p3_b    = (const float*)d_in[21];
  const float* rt_w    = (const float*)d_in[22];
  const float* rt_b    = (const float*)d_in[23];
  const float* it_w    = (const float*)d_in[24];
  const float* it_b    = (const float*)d_in[25];
  const float* msg_wr  = (const float*)d_in[26];
  const float* msg_wi  = (const float*)d_in[27];
  const float* mc1_w   = (const float*)d_in[28];
  const float* mc1_b   = (const float*)d_in[29];
  const float* mc2_w   = (const float*)d_in[30];
  const float* mc2_b   = (const float*)d_in[31];
  const float* pc1_w   = (const float*)d_in[32];
  const float* pc1_b   = (const float*)d_in[33];
  const float* pc2_w   = (const float*)d_in[34];
  const float* pc2_b   = (const float*)d_in[35];

  // ---- workspace carve-up (~54 MB of f32) ----
  size_t off = 0;
  auto alloc = [&](size_t n) -> float* {
    float* p = (float*)d_ws + off;
    off += (n + 63) & ~(size_t)63;
    return p;
  };
  float* H1  = alloc((size_t)N0_ * H_);
  float* O1  = alloc((size_t)N0_ * H_);
  float* DEG = alloc(SORTN);
  float* SC  = alloc(SORTN);
  int*   PERM = (int*)alloc(SORTN);
  int*   MAPg = (int*)alloc(SORTN);
  int*   SRC  = (int*)alloc(EDG);
  int*   DST  = (int*)alloc(EDG);
  float* EM   = alloc(EDG);
  float* XN1 = alloc((size_t)K1_ * H_);
  float* H2  = alloc((size_t)K1_ * H_);
  float* O2  = alloc((size_t)K1_ * H_);
  float* XN2 = alloc((size_t)K2_ * H_);
  float* XRe = alloc((size_t)K2_ * H_);
  float* XIm = alloc((size_t)K2_ * H_);
  float* HE1 = alloc((size_t)K2_ * UH_);
  float* OE1 = alloc((size_t)K2_ * UH_);
  float* HE2 = alloc((size_t)K2_ * UH_);
  float* OE2 = alloc((size_t)K2_ * UH_);
  float* EMB = alloc(UH_);
  float* T1V = alloc(UH_);
  float* T2V = alloc(UH_);
  float* PM  = alloc(K2SQ_);
  float* Xr  = alloc(K2SQ_);
  float* Xi  = alloc(K2SQ_);
  float* Tar = alloc(K2SQ_);
  float* Tai = alloc(K2SQ_);
  float* Tbr = alloc(K2SQ_);
  float* Tbi = alloc(K2SQ_);
  float* XRT = alloc((size_t)K2_ * H_);
  float* XIT = alloc((size_t)K2_ * H_);
  float* EVr = alloc((size_t)K2_ * H_);
  float* EVi = alloc((size_t)K2_ * H_);
  float* TCr = alloc((size_t)K2_ * H_);
  float* TCi = alloc((size_t)K2_ * H_);
  float* AGr = alloc((size_t)K2_ * H_);
  float* AGi = alloc((size_t)K2_ * H_);
  float* MAGb = alloc((size_t)K2_ * H_);
  float* PHb  = alloc((size_t)K2_ * H_);
  float* GM  = alloc(H_);
  float* GP  = alloc(H_);
  float* LM1 = alloc(H_ / 2);
  float* LP1 = alloc(H_ / 2);
  float* LM2 = alloc(16);
  float* LP2 = alloc(16);
  // packed-fragment scratch (v16h = 8 f32 slots)
  v16h* PAr_hi = (v16h*)alloc((size_t)APACK_V * 8);
  v16h* PAr_lo = (v16h*)alloc((size_t)APACK_V * 8);
  v16h* PAi_hi = (v16h*)alloc((size_t)APACK_V * 8);
  v16h* PAi_lo = (v16h*)alloc((size_t)APACK_V * 8);
  v16h* PBr_hi = (v16h*)alloc((size_t)BPACK_V * 8);
  v16h* PBr_lo = (v16h*)alloc((size_t)BPACK_V * 8);
  v16h* PBi_hi = (v16h*)alloc((size_t)BPACK_V * 8);
  v16h* PBi_lo = (v16h*)alloc((size_t)BPACK_V * 8);
  v16h* XAr_hi = (v16h*)alloc((size_t)BPACK_V * 8);   // X in A-layout (41*21*32)
  v16h* XAr_lo = (v16h*)alloc((size_t)BPACK_V * 8);
  v16h* XAi_hi = (v16h*)alloc((size_t)BPACK_V * 8);
  v16h* XAi_lo = (v16h*)alloc((size_t)BPACK_V * 8);

  const int TPB = 256;
  #define GRD(n) dim3(((n) + TPB - 1) / TPB)

  auto packA = [&](const float* A, int lda, int M, int K, v16h* hi, v16h* lo) {
    int MT = (M + 15) / 16, KT = (K + 31) / 32;
    int n = MT * KT * 32;
    k_pack_a<<<GRD(n), TPB, 0, stream>>>(A, lda, M, K, KT, hi, lo, n);
  };
  auto packB = [&](const float* B, int ldb, int K, int N, v16h* hi, v16h* lo) {
    int NT = (N + 15) / 16, KT = (K + 31) / 32;
    int n = NT * KT * 32;
    k_pack_b<<<GRD(n), TPB, 0, stream>>>(B, ldb, K, N, KT, hi, lo, n);
  };
  auto gemm_pk = [&](const v16h* ahi, const v16h* alo, const v16h* bhi, const v16h* blo,
                     float* C, int ldc, int M, int N, int K,
                     float alpha, float beta, const float* bias, int act) {
    dim3 g((N + 15) / 16, (M + 15) / 16);
    k_gemm_pk<<<g, dim3(32), 0, stream>>>(ahi, alo, bhi, blo, C, ldc, M, N,
                                          (K + 31) / 32, alpha, beta, bias, act);
  };
  // LDS/TDM-staged variant: used for all complex GEMMs (KT <= KT_MAX)
  auto gemm_lds = [&](const v16h* ahi, const v16h* alo, const v16h* bhi, const v16h* blo,
                      float* C, int ldc, int M, int N, int K, float alpha, float beta) {
    int NT = (N + 15) / 16, MT = (M + 15) / 16;
    dim3 g((NT + 7) / 8, MT);
    k_gemm_pk_lds<<<g, dim3(256), 0, stream>>>(ahi, alo, bhi, blo, C, ldc, M, N,
                                               (K + 31) / 32, alpha, beta);
  };
  // real GEMM: pack then multiply (per-wave kernel; supports bias/act)
  auto gemm = [&](const float* A, int lda, const float* B, int ldb,
                  float* C, int ldc, int M, int N, int K,
                  float alpha, float beta, const float* bias, int act) {
    packA(A, lda, M, K, PAr_hi, PAr_lo);
    packB(B, ldb, K, N, PBr_hi, PBr_lo);
    gemm_pk(PAr_hi, PAr_lo, PBr_hi, PBr_lo, C, ldc, M, N, K, alpha, beta, bias, act);
  };
  // complex GEMM: C = alpha * (Ar+iAi)(Br+iBi)   (LDS/TDM-staged)
  auto cgemm = [&](float* Cr, float* Ci, const float* Ar, const float* Ai, int lda,
                   const float* Br, const float* Bi, int ldb, int ldc,
                   int M, int N, int K, float alpha) {
    packA(Ar, lda, M, K, PAr_hi, PAr_lo);
    packA(Ai, lda, M, K, PAi_hi, PAi_lo);
    packB(Br, ldb, K, N, PBr_hi, PBr_lo);
    packB(Bi, ldb, K, N, PBi_hi, PBi_lo);
    gemm_lds(PAr_hi, PAr_lo, PBr_hi, PBr_lo, Cr, ldc, M, N, K,  alpha, 0.f);
    gemm_lds(PAi_hi, PAi_lo, PBi_hi, PBi_lo, Cr, ldc, M, N, K, -alpha, 1.f);
    gemm_lds(PAr_hi, PAr_lo, PBi_hi, PBi_lo, Ci, ldc, M, N, K,  alpha, 0.f);
    gemm_lds(PAi_hi, PAi_lo, PBr_hi, PBr_lo, Ci, ldc, M, N, K,  alpha, 1.f);
  };
  // complex GEMM with A = X already packed (Horner loop: X is invariant)
  auto cgemm_x = [&](float* Cr, float* Ci, const float* Br, const float* Bi,
                     int ldb, int ldc, int M, int N, int K, float alpha) {
    packB(Br, ldb, K, N, PBr_hi, PBr_lo);
    packB(Bi, ldb, K, N, PBi_hi, PBi_lo);
    gemm_lds(XAr_hi, XAr_lo, PBr_hi, PBr_lo, Cr, ldc, M, N, K,  alpha, 0.f);
    gemm_lds(XAi_hi, XAi_lo, PBi_hi, PBi_lo, Cr, ldc, M, N, K, -alpha, 1.f);
    gemm_lds(XAr_hi, XAr_lo, PBi_hi, PBi_lo, Ci, ldc, M, N, K,  alpha, 0.f);
    gemm_lds(XAi_hi, XAi_lo, PBr_hi, PBr_lo, Ci, ldc, M, N, K,  alpha, 1.f);
  };

  // ========== GCN layer 1 (4096 nodes, 65536 edges, em = 1) ==========
  gemm(x, DIN_, gcn1_w, H_, H1, H_, N0_, H_, DIN_, 1.f, 0.f, nullptr, 0);
  k_fill<<<GRD(N0_), TPB, 0, stream>>>(DEG, 1.f, N0_);
  k_deg<<<GRD(EDG), TPB, 0, stream>>>(e_dst, nullptr, DEG, EDG);
  k_self_init<<<GRD(N0_ * H_), TPB, 0, stream>>>(H1, DEG, gcn1_b, O1, N0_, H_);
  k_gcn_scatter<<<GRD(EDG * H_), TPB, 0, stream>>>(H1, e_src, e_dst, nullptr, DEG, O1, EDG, H_);
  k_relu<<<GRD(N0_ * H_), TPB, 0, stream>>>(O1, N0_ * H_);

  // ========== TopK pool 1 (keep K1 = 3277) ==========
  k_score<<<GRD(N0_), TPB, 0, stream>>>(O1, pool1_w, SC, N0_, H_);
  k_topk_sort<<<1, 512, 0, stream>>>(SC, N0_, PERM, MAPg, K1_);
  k_gather_scale<<<GRD(K1_ * H_), TPB, 0, stream>>>(O1, PERM, SC, XN1, K1_, H_);
  k_remap<<<GRD(EDG), TPB, 0, stream>>>(e_src, e_dst, nullptr, MAPg, SRC, DST, EM, EDG);

  // ========== GCN layer 2 ==========
  gemm(XN1, H_, gcn2_w, H_, H2, H_, K1_, H_, H_, 1.f, 0.f, nullptr, 0);
  k_fill<<<GRD(K1_), TPB, 0, stream>>>(DEG, 1.f, K1_);
  k_deg<<<GRD(EDG), TPB, 0, stream>>>(DST, EM, DEG, EDG);
  k_self_init<<<GRD(K1_ * H_), TPB, 0, stream>>>(H2, DEG, gcn2_b, O2, K1_, H_);
  k_gcn_scatter<<<GRD(EDG * H_), TPB, 0, stream>>>(H2, SRC, DST, EM, DEG, O2, EDG, H_);
  k_relu<<<GRD(K1_ * H_), TPB, 0, stream>>>(O2, K1_ * H_);

  // ========== TopK pool 2 (keep K2 = 656) ==========
  k_score<<<GRD(K1_), TPB, 0, stream>>>(O2, pool2_w, SC, K1_, H_);
  k_topk_sort<<<1, 512, 0, stream>>>(SC, K1_, PERM, MAPg, K2_);
  k_gather_scale<<<GRD(K2_ * H_), TPB, 0, stream>>>(O2, PERM, SC, XN2, K2_, H_);
  k_remap<<<GRD(EDG), TPB, 0, stream>>>(SRC, DST, EM, MAPg, SRC, DST, EM, EDG);

  // ========== complex projections xr, xi ==========
  gemm(XN2, H_, inr_w, H_, XRe, H_, K2_, H_, H_, 1.f, 0.f, inr_b, 0);
  gemm(XN2, H_, ini_w, H_, XIm, H_, K2_, H_, H_, 1.f, 0.f, ini_b, 0);

  // ========== encoder GCN on ones (xg = 1, enc1_w is 1xUH) ==========
  k_broadcast_row<<<GRD(K2_ * UH_), TPB, 0, stream>>>(enc1_w, HE1, K2_, UH_);
  k_fill<<<GRD(K2_), TPB, 0, stream>>>(DEG, 1.f, K2_);
  k_deg<<<GRD(EDG), TPB, 0, stream>>>(DST, EM, DEG, EDG);
  k_self_init<<<GRD(K2_ * UH_), TPB, 0, stream>>>(HE1, DEG, enc1_b, OE1, K2_, UH_);
  k_gcn_scatter<<<GRD(EDG * UH_), TPB, 0, stream>>>(HE1, SRC, DST, EM, DEG, OE1, EDG, UH_);
  k_relu<<<GRD(K2_ * UH_), TPB, 0, stream>>>(OE1, K2_ * UH_);
  gemm(OE1, UH_, enc2_w, UH_, HE2, UH_, K2_, UH_, UH_, 1.f, 0.f, nullptr, 0);
  k_self_init<<<GRD(K2_ * UH_), TPB, 0, stream>>>(HE2, DEG, enc2_b, OE2, K2_, UH_);
  k_gcn_scatter<<<GRD(EDG * UH_), TPB, 0, stream>>>(HE2, SRC, DST, EM, DEG, OE2, EDG, UH_);
  k_colmean<<<GRD(UH_), TPB, 0, stream>>>(OE2, EMB, K2_, UH_);

  // ========== MLP: emb -> pm (tanh) ==========
  k_vecmat<<<GRD(UH_), TPB, 0, stream>>>(EMB, p1_w, p1_b, T1V, UH_, UH_, 1);
  k_vecmat<<<GRD(UH2_), TPB, 0, stream>>>(T1V, p2_w, p2_b, T2V, UH_, UH2_, 1);
  k_vecmat<<<GRD(K2SQ_), TPB, 0, stream>>>(T2V, p3_w, p3_b, PM, UH2_, K2SQ_, 2);

  // ========== expm(A) by scaling-and-squaring + order-12 Horner Taylor ====
  // X = A / 2^s; A = 0.05*(0.5*(pm-pm^T) + i*0.5*(pm+pm^T))
  const float inv2s = 1.f / (float)(1 << EXPM_S);
  k_buildX<<<GRD(K2SQ_), TPB, 0, stream>>>(PM, Xr, Xi, K2_, 0.05f * 0.5f * inv2s);
  packA(Xr, K2_, K2_, K2_, XAr_hi, XAr_lo);      // pack X once for all 11 iters
  packA(Xi, K2_, K2_, K2_, XAi_hi, XAi_lo);
  float *Tr = Tar, *Ti = Tai, *Ur = Tbr, *Ui = Tbi;
  k_scale<<<GRD(K2SQ_), TPB, 0, stream>>>(Xr, Tr, 1.f / (float)EXPM_M, K2SQ_);
  k_scale<<<GRD(K2SQ_), TPB, 0, stream>>>(Xi, Ti, 1.f / (float)EXPM_M, K2SQ_);
  for (int k = EXPM_M - 1; k >= 1; --k) {
    k_add_diag<<<GRD(K2_), TPB, 0, stream>>>(Tr, K2_, 1.f);      // B = I + T
    cgemm_x(Ur, Ui, Tr, Ti, K2_, K2_, K2_, K2_, K2_, 1.f / (float)k);
    float* t;
    t = Tr; Tr = Ur; Ur = t;
    t = Ti; Ti = Ui; Ui = t;
  }
  k_add_diag<<<GRD(K2_), TPB, 0, stream>>>(Tr, K2_, 1.f);        // E = I + T
  for (int s = 0; s < EXPM_S; ++s) {                             // U <- U*U
    cgemm(Ur, Ui, Tr, Ti, K2_, Tr, Ti, K2_, K2_, K2_, K2_, K2_, 1.f);
    float* t;
    t = Tr; Tr = Ur; Ur = t;
    t = Ti; Ti = Ui; Ui = t;
  }
  // U = (Tr, Ti). Note: herm_sqrt/eigh are dead code -- U_dil[:K2,:K2] == U.

  // ========== xc = relu(xr@rt_w+b) + i*relu(xi@it_w+b); evolved = U @ xc ==
  gemm(XRe, H_, rt_w, H_, XRT, H_, K2_, H_, H_, 1.f, 0.f, rt_b, 1);
  gemm(XIm, H_, it_w, H_, XIT, H_, K2_, H_, H_, 1.f, 0.f, it_b, 1);
  cgemm(EVr, EVi, Tr, Ti, K2_, XRT, XIT, H_, H_, K2_, H_, K2_, 1.f);

  // ========== message passing, reassociated:
  //   agg = (scatter_add(em * evolved[src])) @ Wm
  k_fill<<<GRD(K2_ * H_), TPB, 0, stream>>>(TCr, 0.f, K2_ * H_);
  k_fill<<<GRD(K2_ * H_), TPB, 0, stream>>>(TCi, 0.f, K2_ * H_);
  k_scatter_plain<<<GRD(EDG * H_), TPB, 0, stream>>>(EVr, SRC, DST, EM, TCr, EDG, H_);
  k_scatter_plain<<<GRD(EDG * H_), TPB, 0, stream>>>(EVi, SRC, DST, EM, TCi, EDG, H_);
  cgemm(AGr, AGi, TCr, TCi, H_, msg_wr, msg_wi, H_, H_, K2_, H_, H_, 1.f);

  // ========== magnitude/phase heads ==========
  k_polar<<<GRD(K2_ * H_), TPB, 0, stream>>>(AGr, AGi, XRe, XIm, MAGb, PHb, K2_ * H_);
  k_colmean<<<GRD(H_), TPB, 0, stream>>>(MAGb, GM, K2_, H_);
  k_colmean<<<GRD(H_), TPB, 0, stream>>>(PHb, GP, K2_, H_);
  k_vecmat<<<GRD(H_ / 2), TPB, 0, stream>>>(GM, mc1_w, mc1_b, LM1, H_, H_ / 2, 1);
  k_vecmat<<<GRD(NC_), TPB, 0, stream>>>(LM1, mc2_w, mc2_b, LM2, H_ / 2, NC_, 0);
  k_vecmat<<<GRD(H_ / 2), TPB, 0, stream>>>(GP, pc1_w, pc1_b, LP1, H_, H_ / 2, 1);
  k_vecmat<<<GRD(NC_), TPB, 0, stream>>>(LP1, pc2_w, pc2_b, LP2, H_ / 2, NC_, 0);
  k_addvec<<<GRD(NC_), TPB, 0, stream>>>(LM2, LP2, (float*)d_out, NC_);
  #undef GRD
}
